// RotaryCrossAttention_1494648619013
// MI455X (gfx1250) — compile-verified, matrix operand order checked by reading, not run on hardware
//
#include <hip/hip_runtime.h>
#include <hip/hip_bf16.h>
#include <stdint.h>

// ---------------------------------------------------------------------------
// RotaryCrossAttention for MI455X (gfx1250, wave32, WMMA)
// B=2, NQ=1024, NK=8192, D=512, H=8, DH=64, INNER=512
// Matmuls: v_wmma_f32_16x16x32_bf16.  GEMMs software-pipelined (next k-step
// fragments in flight during current WMMAs -> partial s_wait_loadcnt).
// Attention K/V staged to LDS via global_load_async_to_lds_b128 (ASYNCcnt),
// double-buffered per 32-key chunk.
// ---------------------------------------------------------------------------

typedef __attribute__((ext_vector_type(16))) __bf16 bf16x16;
typedef __attribute__((ext_vector_type(8)))  float  f32x8;

struct Pack32 { uint4 lo; uint4 hi; };   // 32 bytes == bf16x16

__device__ __forceinline__ unsigned short f2bfu(float f) {
  unsigned u = __builtin_bit_cast(unsigned, f);
  u += 0x7FFFu + ((u >> 16) & 1u);       // round-to-nearest-even
  return (unsigned short)(u >> 16);
}

// Load one 16x32 bf16 WMMA fragment.  Per lane: two contiguous 16-byte loads
// from row `p`, at element offsets hk and 16+hk where hk = (lane>>4)*8.
__device__ __forceinline__ bf16x16 ld_frag(const unsigned short* p, int hk) {
  Pack32 r;
  r.lo = *(const uint4*)(p + hk);
  r.hi = *(const uint4*)(p + hk + 16);
  return __builtin_bit_cast(bf16x16, r);
}

__device__ __forceinline__ f32x8 wmma_bf16(bf16x16 a, bf16x16 b, f32x8 c) {
  return __builtin_amdgcn_wmma_f32_16x16x32_bf16(
      false, a, false, b, (short)0, c, false, false);
}

// 16-byte async copy global -> LDS (per lane), tracked by ASYNCcnt.
__device__ __forceinline__ void async_load16(uint32_t lds_byte, uint64_t gaddr) {
  asm volatile("global_load_async_to_lds_b128 %0, %1, off"
               :: "v"(lds_byte), "v"(gaddr) : "memory");
}
__device__ __forceinline__ void wait_async0() {
  asm volatile("s_wait_asynccnt 0x0" ::: "memory");
}
// Generic pointer to a __shared__ object: low 32 bits == LDS byte offset
// (flat->LDS aperture truncates to addr[31:0]).
__device__ __forceinline__ uint32_t lds_off(const void* p) {
  return (uint32_t)(uintptr_t)p;
}

// ---------------------------------------------------------------------------
// 1) LayerNorm (D=512) -> bf16.  One block (256 thr) per row.
// ---------------------------------------------------------------------------
__global__ __launch_bounds__(256) void ln_bf16_kernel(
    const float* __restrict__ x, const float* __restrict__ w,
    const float* __restrict__ bvec, unsigned short* __restrict__ out) {
  const int D = 512;
  int row = blockIdx.x;
  int t = threadIdx.x;
  const float* xr = x + (size_t)row * D;
  float a0 = xr[t];
  float a1 = xr[t + 256];
  __shared__ float sred[8];
  float s = a0 + a1;
  for (int off = 16; off >= 1; off >>= 1) s += __shfl_xor(s, off, 32);
  if ((t & 31) == 0) sred[t >> 5] = s;
  __syncthreads();
  float mean = 0.f;
  for (int i = 0; i < 8; ++i) mean += sred[i];
  mean *= (1.f / D);
  float d0 = a0 - mean, d1 = a1 - mean;
  float vs = d0 * d0 + d1 * d1;
  for (int off = 16; off >= 1; off >>= 1) vs += __shfl_xor(vs, off, 32);
  __syncthreads();
  if ((t & 31) == 0) sred[t >> 5] = vs;
  __syncthreads();
  float var = 0.f;
  for (int i = 0; i < 8; ++i) var += sred[i];
  var *= (1.f / D);
  float r = rsqrtf(var + 1e-5f);
  unsigned short* orow = out + (size_t)row * D;
  orow[t]       = f2bfu(d0 * r * w[t] + bvec[t]);
  orow[t + 256] = f2bfu(d1 * r * w[t + 256] + bvec[t + 256]);
}

// ---------------------------------------------------------------------------
// 2) f32 [K][N] weight -> bf16 transposed [N][K]
// ---------------------------------------------------------------------------
__global__ __launch_bounds__(256) void cvt_transpose_kernel(
    const float* __restrict__ W, unsigned short* __restrict__ Wt, int K, int N) {
  int idx = blockIdx.x * 256 + threadIdx.x;
  if (idx >= K * N) return;
  int k = idx / N, n = idx - k * N;
  Wt[(size_t)n * K + k] = f2bfu(W[idx]);
}

// ---------------------------------------------------------------------------
// Wave-level GEMM tile: 16(M) x 64(N), K-loop in steps of 32, one-stage
// software pipeline: iteration k's WMMAs run while k+32's 10 b128 loads are
// in flight (partial s_wait_loadcnt instead of a full drain).
// X: [M][K] bf16 row-major,  Wt: [N][K] bf16 row-major (transposed weight).
// ---------------------------------------------------------------------------
__device__ __forceinline__ void gemm_tile(
    const unsigned short* __restrict__ X, const unsigned short* __restrict__ Wt,
    int K, int mbase, int nbase, int lane, f32x8 acc[4]) {
  int hk = (lane >> 4) * 8;
  int ln = lane & 15;
  const unsigned short* xr = X + (size_t)(mbase + ln) * K;
  const unsigned short* w0 = Wt + (size_t)(nbase + 0  + ln) * K;
  const unsigned short* w1 = Wt + (size_t)(nbase + 16 + ln) * K;
  const unsigned short* w2 = Wt + (size_t)(nbase + 32 + ln) * K;
  const unsigned short* w3 = Wt + (size_t)(nbase + 48 + ln) * K;

  bf16x16 a  = ld_frag(xr, hk);
  bf16x16 b0 = ld_frag(w0, hk);
  bf16x16 b1 = ld_frag(w1, hk);
  bf16x16 b2 = ld_frag(w2, hk);
  bf16x16 b3 = ld_frag(w3, hk);
  for (int k0 = 32; k0 < K; k0 += 32) {
    bf16x16 an  = ld_frag(xr + k0, hk);   // next-stage loads issued first
    bf16x16 bn0 = ld_frag(w0 + k0, hk);
    bf16x16 bn1 = ld_frag(w1 + k0, hk);
    bf16x16 bn2 = ld_frag(w2 + k0, hk);
    bf16x16 bn3 = ld_frag(w3 + k0, hk);
    acc[0] = wmma_bf16(a, b0, acc[0]);    // consume previous stage
    acc[1] = wmma_bf16(a, b1, acc[1]);
    acc[2] = wmma_bf16(a, b2, acc[2]);
    acc[3] = wmma_bf16(a, b3, acc[3]);
    a = an; b0 = bn0; b1 = bn1; b2 = bn2; b3 = bn3;
  }
  acc[0] = wmma_bf16(a, b0, acc[0]);      // pipeline epilogue
  acc[1] = wmma_bf16(a, b1, acc[1]);
  acc[2] = wmma_bf16(a, b2, acc[2]);
  acc[3] = wmma_bf16(a, b3, acc[3]);
}

// ---------------------------------------------------------------------------
// 3) q = LN(xq) @ Wq, fused RoPE -> q_rope [B,H,NQ,DH] bf16.  M=2048,N=512,K=512
// ---------------------------------------------------------------------------
__global__ __launch_bounds__(256) void qproj_rope_kernel(
    const unsigned short* __restrict__ X, const unsigned short* __restrict__ Wt,
    const float* __restrict__ ropeq, unsigned short* __restrict__ qout) {
  const int K = 512, NQ = 1024;
  int wid  = (blockIdx.x * 256 + threadIdx.x) >> 5;
  int lane = threadIdx.x & 31;
  int tm = wid & 127, tn = wid >> 7;
  int mbase = tm * 16, nbase = tn * 64;
  f32x8 acc[4];
  for (int t = 0; t < 4; ++t) for (int i = 0; i < 8; ++i) acc[t][i] = 0.f;
  gemm_tile(X, Wt, K, mbase, nbase, lane, acc);
  int ln = lane & 15, rb = (lane >> 4) * 8;
  for (int t = 0; t < 4; ++t) {
    int col = nbase + t * 16 + ln;
    int h = col >> 6, dh = col & 63;
    for (int i = 0; i < 8; ++i) {
      int gm = mbase + rb + i;
      int b = gm >> 10, nq = gm & 1023;
      float v  = acc[t][i];
      float pv = __shfl_xor(v, 1, 32);          // partner (dh^1), same row
      float f  = ropeq[((size_t)b * NQ + nq) * 64 + dh];
      float cs = __cosf(f), sn = __sinf(f);
      float y  = (col & 1) ? (v * cs + pv * sn) : (v * cs - pv * sn);
      qout[(((size_t)(b * 8 + h) * NQ + nq) * 64 + dh)] = f2bfu(y);
    }
  }
}

// ---------------------------------------------------------------------------
// 4) kv = LN(xc) @ Wkv, fused RoPE on K and V.
//    K -> [B,H,NK,DH]; V -> transposed [B,H,DH,NK].  M=16384,N=1024,K=512
// ---------------------------------------------------------------------------
__global__ __launch_bounds__(256) void kvproj_rope_kernel(
    const unsigned short* __restrict__ X, const unsigned short* __restrict__ Wt,
    const float* __restrict__ ropek, unsigned short* __restrict__ kout,
    unsigned short* __restrict__ vtout) {
  const int K = 512, NK = 8192;
  int wid  = (blockIdx.x * 256 + threadIdx.x) >> 5;
  int lane = threadIdx.x & 31;
  int tm = wid & 1023, tn = wid >> 10;
  int mbase = tm * 16, nbase = tn * 64;
  f32x8 acc[4];
  for (int t = 0; t < 4; ++t) for (int i = 0; i < 8; ++i) acc[t][i] = 0.f;
  gemm_tile(X, Wt, K, mbase, nbase, lane, acc);
  int ln = lane & 15, rb = (lane >> 4) * 8;
  for (int t = 0; t < 4; ++t) {
    int col = nbase + t * 16 + ln;
    int isv = col >> 9;               // 0 = K half, 1 = V half
    int c = col & 511;
    int h = c >> 6, dh = c & 63;
    for (int i = 0; i < 8; ++i) {
      int gm = mbase + rb + i;
      int b = gm >> 13, nk = gm & 8191;
      float v  = acc[t][i];
      float pv = __shfl_xor(v, 1, 32);
      float f  = ropek[((size_t)b * NK + nk) * 64 + dh];
      float cs = __cosf(f), sn = __sinf(f);
      float y  = (col & 1) ? (v * cs + pv * sn) : (v * cs - pv * sn);
      unsigned short yb = f2bfu(y);
      if (isv) vtout[((size_t)(b * 8 + h) * 64 + dh) * NK + nk] = yb;
      else     kout[(((size_t)(b * 8 + h) * NK + nk) * 64 + dh)] = yb;
    }
  }
}

// ---------------------------------------------------------------------------
// 5) Flash attention.  8 waves/block share one (b,h); per 32-key chunk wave 0
//    stages K (32x64) and V^T (64x32) into LDS with async b128 copies
//    (double-buffered); every wave runs 4 WMMA (S) + online softmax + LDS P
//    bounce + 4 WMMA (P.V) out of LDS.
// ---------------------------------------------------------------------------
__device__ __forceinline__ void stage_kv_async(
    const unsigned short* kbase, const unsigned short* vbase,
    int kb, int lane, uint32_t ktoff, uint32_t vtoff) {
  const int NK = 8192;
  for (int j = 0; j < 8; ++j) {          // K tile: 32 rows x 128B (8 x16B/row)
    int lin = j * 32 + lane;
    int row = lin >> 3, off = lin & 7;
    uint64_t g = (uint64_t)(uintptr_t)(kbase + (size_t)(kb + row) * 64 + off * 8);
    async_load16(ktoff + (uint32_t)(row * 128 + off * 16), g);
  }
  for (int j = 0; j < 8; ++j) {          // V^T tile: 64 rows x 64B (4 x16B/row)
    int lin = j * 32 + lane;
    int row = lin >> 2, off = lin & 3;
    uint64_t g = (uint64_t)(uintptr_t)(vbase + (size_t)row * NK + kb + off * 8);
    async_load16(vtoff + (uint32_t)(row * 64 + off * 16), g);
  }
}

__global__ __launch_bounds__(256) void attn_kernel(
    const unsigned short* __restrict__ qbuf, const unsigned short* __restrict__ kbuf,
    const unsigned short* __restrict__ vtbuf, const unsigned char* __restrict__ mask,
    const float* __restrict__ ropeq, unsigned short* __restrict__ aout) {
  const int NQ = 1024, NK = 8192, H = 8;
  __shared__ unsigned short kt[2][32 * 64];    // K chunk: [key][dh]
  __shared__ unsigned short vt[2][64 * 32];    // V^T chunk: [dh][key]
  __shared__ unsigned short pbuf[8][16 * 32];  // per-wave P tile

  int lane = threadIdx.x & 31;
  int wib  = threadIdx.x >> 5;
  int bh = blockIdx.x >> 3;            // 16 (b,h) pairs
  int qg = blockIdx.x & 7;             // 8 q-groups of 8 q-tiles
  int b = bh >> 3, h = bh & 7;
  int qt = qg * 8 + wib;
  int hk = (lane >> 4) * 8;
  int ln = lane & 15;
  int rb = (lane >> 4) * 8;

  const unsigned short* qrow = qbuf + ((size_t)(b * H + h) * NQ + qt * 16 + ln) * 64;
  bf16x16 aq0 = ld_frag(qrow, hk);
  bf16x16 aq1 = ld_frag(qrow + 32, hk);

  f32x8 o[4];
  float rowmax[8], rowsum[8];
  for (int t = 0; t < 4; ++t) for (int i = 0; i < 8; ++i) o[t][i] = 0.f;
  for (int i = 0; i < 8; ++i) { rowmax[i] = -1e30f; rowsum[i] = 0.f; }

  const unsigned short* kbase = kbuf  + (size_t)(b * H + h) * NK * 64;
  const unsigned short* vbase = vtbuf + (size_t)(b * H + h) * 64 * NK;
  const unsigned char*  mrow  = mask  + (size_t)b * NK;
  unsigned short* pw = pbuf[wib];
  uint32_t ktoff[2] = { lds_off(&kt[0][0]), lds_off(&kt[1][0]) };
  uint32_t vtoff[2] = { lds_off(&vt[0][0]), lds_off(&vt[1][0]) };

  // Prologue: fill buffer 0.
  if (wib == 0) {
    stage_kv_async(kbase, vbase, 0, lane, ktoff[0], vtoff[0]);
    wait_async0();
  }
  __syncthreads();

  int buf = 0;
  for (int kb = 0; kb < NK; kb += 32) {
    // Kick off the next chunk while this one is consumed.
    if (wib == 0 && kb + 32 < NK)
      stage_kv_async(kbase, vbase, kb + 32, lane, ktoff[buf ^ 1], vtoff[buf ^ 1]);

    const unsigned short* ktile = kt[buf];
    const unsigned short* vtile = vt[buf];

    // ---- S = Q.K^T (two 16-key n-tiles) ----
    f32x8 s0, s1;
    for (int i = 0; i < 8; ++i) { s0[i] = 0.f; s1[i] = 0.f; }
    const unsigned short* k0r = ktile + (size_t)ln * 64;
    const unsigned short* k1r = ktile + (size_t)(16 + ln) * 64;
    s0 = wmma_bf16(aq0, ld_frag(k0r,      hk), s0);
    s0 = wmma_bf16(aq1, ld_frag(k0r + 32, hk), s0);
    s1 = wmma_bf16(aq0, ld_frag(k1r,      hk), s1);
    s1 = wmma_bf16(aq1, ld_frag(k1r + 32, hk), s1);

    float mb0 = mrow[kb + ln]      ? 0.f : -1e30f;
    float mb1 = mrow[kb + 16 + ln] ? 0.f : -1e30f;

    // ---- online softmax (row fixed per lane-group; xor over lanes 0..15) ----
    float p0[8], p1[8];
    for (int i = 0; i < 8; ++i) {
      float a0 = s0[i] * 0.125f + mb0;   // scale = DH^-0.5
      float a1 = s1[i] * 0.125f + mb1;
      float rm = fmaxf(a0, a1);
      for (int off = 1; off < 16; off <<= 1) rm = fmaxf(rm, __shfl_xor(rm, off, 32));
      float nm = fmaxf(rowmax[i], rm);
      float alpha = __expf(rowmax[i] - nm);
      rowmax[i] = nm;
      float e0 = __expf(a0 - nm), e1 = __expf(a1 - nm);
      float ps = e0 + e1;
      for (int off = 1; off < 16; off <<= 1) ps += __shfl_xor(ps, off, 32);
      rowsum[i] = rowsum[i] * alpha + ps;
      for (int t = 0; t < 4; ++t) o[t][i] *= alpha;
      p0[i] = e0; p1[i] = e1;
    }

    // ---- P: C-layout regs -> LDS 16x32 -> A-layout fragment (same wave) ----
    for (int i = 0; i < 8; ++i) {
      pw[(rb + i) * 32 + ln]      = f2bfu(p0[i]);
      pw[(rb + i) * 32 + 16 + ln] = f2bfu(p1[i]);
    }
    bf16x16 ap = ld_frag(pw + ln * 32, hk);   // same-wave DS ops are in-order

    // ---- O += P.V from staged V^T ----
    for (int t = 0; t < 4; ++t)
      o[t] = wmma_bf16(ap, ld_frag(vtile + (size_t)(t * 16 + ln) * 32, hk), o[t]);

    __syncthreads();                     // all readers done with `buf`
    if (wib == 0 && kb + 32 < NK) wait_async0();
    __syncthreads();                     // next buffer published
    buf ^= 1;
  }

  // ---- epilogue: normalize, inverse RoPE (-freqs), store ----
  for (int i = 0; i < 8; ++i) {
    float inv = 1.f / rowsum[i];
    for (int t = 0; t < 4; ++t) o[t][i] *= inv;
  }
  for (int t = 0; t < 4; ++t) {
    int dh = t * 16 + ln;
    for (int i = 0; i < 8; ++i) {
      int nq = qt * 16 + rb + i;
      float v  = o[t][i];
      float pv = __shfl_xor(v, 1, 32);
      float f  = ropeq[((size_t)b * NQ + nq) * 64 + dh];
      float cs = __cosf(f), sn = __sinf(f);
      float y = (dh & 1) ? (v * cs - pv * sn) : (v * cs + pv * sn);
      aout[((size_t)b * NQ + nq) * 512 + h * 64 + dh] = f2bfu(y);
    }
  }
}

// ---------------------------------------------------------------------------
// 6) out = attn_out @ Wout + bout -> f32 d_out.  M=2048, N=512, K=512.
// ---------------------------------------------------------------------------
__global__ __launch_bounds__(256) void outproj_kernel(
    const unsigned short* __restrict__ X, const unsigned short* __restrict__ Wt,
    const float* __restrict__ bout, float* __restrict__ out) {
  const int K = 512;
  int wid  = (blockIdx.x * 256 + threadIdx.x) >> 5;
  int lane = threadIdx.x & 31;
  int tm = wid & 127, tn = wid >> 7;
  int mbase = tm * 16, nbase = tn * 64;
  f32x8 acc[4];
  for (int t = 0; t < 4; ++t) for (int i = 0; i < 8; ++i) acc[t][i] = 0.f;
  gemm_tile(X, Wt, K, mbase, nbase, lane, acc);
  int ln = lane & 15, rb = (lane >> 4) * 8;
  for (int t = 0; t < 4; ++t) {
    int col = nbase + t * 16 + ln;
    float bb = bout[col];
    for (int i = 0; i < 8; ++i) {
      int gm = mbase + rb + i;
      out[(size_t)gm * 512 + col] = acc[t][i] + bb;
    }
  }
}

// ---------------------------------------------------------------------------
extern "C" void kernel_launch(void* const* d_in, const int* in_sizes, int n_in,
                              void* d_out, int out_size, void* d_ws, size_t ws_size,
                              hipStream_t stream) {
  (void)in_sizes; (void)n_in; (void)out_size; (void)ws_size;
  const float* x_query   = (const float*)d_in[0];
  const float* x_context = (const float*)d_in[1];
  const float* rope_q    = (const float*)d_in[2];
  const float* rope_k    = (const float*)d_in[3];
  const unsigned char* attn_mask = (const unsigned char*)d_in[4];
  const float* ln_q_w = (const float*)d_in[5];
  const float* ln_q_b = (const float*)d_in[6];
  const float* ln_c_w = (const float*)d_in[7];
  const float* ln_c_b = (const float*)d_in[8];
  const float* Wq     = (const float*)d_in[9];
  const float* Wkv    = (const float*)d_in[10];
  const float* Wout   = (const float*)d_in[11];
  const float* bout   = (const float*)d_in[12];

  char* ws = (char*)d_ws;
  const size_t MB = 1u << 20;
  unsigned short* xq_ln = (unsigned short*)(ws + 0 * MB);   //  2 MB: [2048][512]
  unsigned short* xc_ln = (unsigned short*)(ws + 2 * MB);   // 16 MB: [16384][512]
  unsigned short* Wqt   = (unsigned short*)(ws + 18 * MB);  // .5 MB: [512][512]
  unsigned short* Wkvt  = (unsigned short*)(ws + 19 * MB);  //  1 MB: [1024][512]
  unsigned short* Woutt = (unsigned short*)(ws + 20 * MB);  // .5 MB: [512][512]
  unsigned short* qr    = (unsigned short*)(ws + 21 * MB);  //  2 MB: [B,H,NQ,64]
  unsigned short* kr    = (unsigned short*)(ws + 23 * MB);  // 16 MB: [B,H,NK,64]
  unsigned short* vt    = (unsigned short*)(ws + 39 * MB);  // 16 MB: [B,H,64,NK]
  unsigned short* ao    = (unsigned short*)(ws + 55 * MB);  //  2 MB: [2048][512]

  ln_bf16_kernel<<<2048, 256, 0, stream>>>(x_query, ln_q_w, ln_q_b, xq_ln);
  ln_bf16_kernel<<<16384, 256, 0, stream>>>(x_context, ln_c_w, ln_c_b, xc_ln);
  cvt_transpose_kernel<<<(512 * 512 + 255) / 256, 256, 0, stream>>>(Wq, Wqt, 512, 512);
  cvt_transpose_kernel<<<(512 * 1024 + 255) / 256, 256, 0, stream>>>(Wkv, Wkvt, 512, 1024);
  cvt_transpose_kernel<<<(512 * 512 + 255) / 256, 256, 0, stream>>>(Wout, Woutt, 512, 512);
  qproj_rope_kernel<<<128, 256, 0, stream>>>(xq_ln, Wqt, rope_q, qr);
  kvproj_rope_kernel<<<2048, 256, 0, stream>>>(xc_ln, Wkvt, rope_k, kr, vt);
  attn_kernel<<<128, 256, 0, stream>>>(qr, kr, vt, attn_mask, rope_q, ao);
  outproj_kernel<<<128, 256, 0, stream>>>(ao, Woutt, bout, (float*)d_out);
}
